// MessageWeightingFunction_74689481277974
// MI455X (gfx1250) — compile-verified, matrix operand order checked by reading, not run on hardware
//
#include <hip/hip_runtime.h>
#include <hip/hip_bf16.h>

// Problem constants (fixed by the reference file)
#define NTOK 262144
#define NSEG 4096
#define DDIM 256
#define ADIM 128

typedef __attribute__((ext_vector_type(16))) __bf16 v16bf;
typedef __attribute__((ext_vector_type(8)))  float  v8f;

union FragU { v16bf v; uint4 u[2]; };

// ---------------------------------------------------------------------------
// Segment boundary discovery: block counts -> exclusive scan -> ends[] ranks
// ---------------------------------------------------------------------------
__global__ void k_count(const int* __restrict__ q, int* __restrict__ bc) {
    __shared__ int wsum[8];
    int i = blockIdx.x * 256 + threadIdx.x;
    int p = q[i];
    unsigned long long m = __ballot(p);          // wave32: low 32 bits valid
    int lane = threadIdx.x & 31, wv = threadIdx.x >> 5;
    if (lane == 0) wsum[wv] = __popcll(m);
    __syncthreads();
    if (threadIdx.x == 0) {
        int s = 0;
        for (int j = 0; j < 8; ++j) s += wsum[j];
        bc[blockIdx.x] = s;
    }
}

__global__ void k_scan(const int* __restrict__ bc, int* __restrict__ boff) {
    // 1024 values, trivial serial exclusive scan
    int s = 0;
    for (int j = 0; j < 1024; ++j) { boff[j] = s; s += bc[j]; }
}

__global__ void k_ends(const int* __restrict__ q, const int* __restrict__ boff,
                       int* __restrict__ ends) {
    __shared__ int wtot[8], woff[8];
    int i = blockIdx.x * 256 + threadIdx.x;
    int p = q[i];
    unsigned long long m = __ballot(p);
    int lane = threadIdx.x & 31, wv = threadIdx.x >> 5;
    int rankw = __popcll(m & ((1ull << lane) - 1ull));
    if (lane == 0) wtot[wv] = __popcll(m);
    __syncthreads();
    if (threadIdx.x == 0) {
        int s = 0;
        for (int j = 0; j < 8; ++j) { woff[j] = s; s += wtot[j]; }
    }
    __syncthreads();
    if (p) ends[boff[blockIdx.x] + woff[wv] + rankw] = i;
}

// ---------------------------------------------------------------------------
// Weight prep: Wq -> bf16 row-major [A,D]; Wk -> bf16 transposed [D,A]
// (both arranged so WMMA B-fragments read contiguous in K)
// ---------------------------------------------------------------------------
__global__ void k_prepw(const float* __restrict__ Wq, const float* __restrict__ Wk,
                        __bf16* __restrict__ Wq_bf, __bf16* __restrict__ WkT_bf) {
    int a = blockIdx.x;        // 0..127
    int d = threadIdx.x;       // 0..255
    Wq_bf[a * DDIM + d]  = (__bf16)Wq[a * DDIM + d];
    WkT_bf[d * ADIM + a] = (__bf16)Wk[a * DDIM + d];
}

// Gather segment-end rows, converting f32 -> bf16
__global__ void k_gather(const float* __restrict__ X, const int* __restrict__ ends,
                         __bf16* __restrict__ Xe) {
    int b = blockIdx.x;        // 0..4095
    int d = threadIdx.x;       // 0..255
    Xe[b * DDIM + d] = (__bf16)X[(size_t)ends[b] * DDIM + d];
}

// ---------------------------------------------------------------------------
// WMMA fragment loader.
// ISA 7.12.2 16-bit A layout: lane&15 = row, lane>>4 selects K halves:
//   elems 0..7  -> K = kb+0..kb+7        (kb = 0 or 8)
//   elems 8..15 -> K = kb+16..kb+23
// Source must have the 32 K-values contiguous per row (ld = row stride).
// ---------------------------------------------------------------------------
__device__ __forceinline__ v16bf load_frag(const __bf16* __restrict__ base,
                                           int ld, int lane) {
    int r  = lane & 15;
    int kb = (lane >> 4) << 3;          // 0 or 8
    const __bf16* p = base + r * ld + kb;
    FragU f;
    f.u[0] = *(const uint4*)(p);
    f.u[1] = *(const uint4*)(p + 16);
    return f.v;
}

// GEMM1: Q[4096,128] = Xe[4096,256] @ WqT ; one wave per 16x16 tile, K-loop 8
__global__ void k_gemm_q(const __bf16* __restrict__ Xe,
                         const __bf16* __restrict__ Wq_bf,
                         __bf16* __restrict__ Qout) {
    int wv = threadIdx.x >> 5, lane = threadIdx.x & 31;
    int t  = blockIdx.x * 8 + wv;       // 2048 tiles
    int tm = t >> 3;                    // 0..255 (row tiles)
    int tn = t & 7;                     // 0..7   (col tiles, A/16)
    v8f acc = {};
    const __bf16* abase = Xe    + (size_t)tm * 16 * DDIM;
    const __bf16* bbase = Wq_bf + (size_t)tn * 16 * DDIM;   // B[k][n] = Wq[n][k]
    for (int k0 = 0; k0 < DDIM; k0 += 32) {
        v16bf a = load_frag(abase + k0, DDIM, lane);
        v16bf b = load_frag(bbase + k0, DDIM, lane);
        acc = __builtin_amdgcn_wmma_f32_16x16x32_bf16(
                  false, a, false, b, (short)0, acc, false, false);
    }
    int col = lane & 15, ro = (lane >> 4) << 3;
    for (int r = 0; r < 8; ++r)
        Qout[(size_t)(tm * 16 + ro + r) * ADIM + tn * 16 + col] = (__bf16)acc[r];
}

// GEMM2: Vseg[4096,256] = Q[4096,128] @ Wk ; B[k=a][n=d] = WkT[d][a]
__global__ void k_gemm_v(const __bf16* __restrict__ Qin,
                         const __bf16* __restrict__ WkT_bf,
                         float* __restrict__ Vseg) {
    int wv = threadIdx.x >> 5, lane = threadIdx.x & 31;
    int t  = blockIdx.x * 8 + wv;       // 4096 tiles
    int tm = t >> 4;                    // 0..255
    int tn = t & 15;                    // 0..15 (D/16)
    v8f acc = {};
    const __bf16* abase = Qin    + (size_t)tm * 16 * ADIM;
    const __bf16* bbase = WkT_bf + (size_t)tn * 16 * ADIM;
    for (int k0 = 0; k0 < ADIM; k0 += 32) {
        v16bf a = load_frag(abase + k0, ADIM, lane);
        v16bf b = load_frag(bbase + k0, ADIM, lane);
        acc = __builtin_amdgcn_wmma_f32_16x16x32_bf16(
                  false, a, false, b, (short)0, acc, false, false);
    }
    int col = lane & 15, ro = (lane >> 4) << 3;
    for (int r = 0; r < 8; ++r)
        Vseg[(size_t)(tm * 16 + ro + r) * DDIM + tn * 16 + col] = acc[r];
}

// ---------------------------------------------------------------------------
// Streaming pass: one wave per ragged segment.
//   pass1: logits[n] = x[n] . vseg[r]   (coalesced 1KB row reads, f32)
//   pass2: sum of exp(logit - max)
//   pass3: out[n] = exp(logit - max) / denom
// ---------------------------------------------------------------------------
__global__ void k_attend(const float* __restrict__ X,
                         const int* __restrict__ ends,
                         const float* __restrict__ Vseg,
                         float* __restrict__ logits,
                         float* __restrict__ out) {
    int wv = threadIdx.x >> 5, lane = threadIdx.x & 31;
    int r  = blockIdx.x * 8 + wv;       // segment id, 0..4095
    int start = (r == 0) ? 0 : ends[r - 1] + 1;
    int endi  = ends[r];

    // this lane's 8-element slice of the segment query vector
    const float4* vp = (const float4*)(Vseg + (size_t)r * DDIM + lane * 8);
    float4 v0 = vp[0], v1 = vp[1];

    float m = -INFINITY;
    for (int n = start; n <= endi; ++n) {
        const float* row = X + (size_t)n * DDIM + lane * 8;
        __builtin_prefetch(row + 4 * DDIM, 0, 1);   // global_prefetch_b8
        float4 x0 = ((const float4*)row)[0];
        float4 x1 = ((const float4*)row)[1];
        float p = x0.x * v0.x + x0.y * v0.y + x0.z * v0.z + x0.w * v0.w
                + x1.x * v1.x + x1.y * v1.y + x1.z * v1.z + x1.w * v1.w;
        #pragma unroll
        for (int off = 16; off > 0; off >>= 1)
            p += __shfl_xor(p, off, 32);
        if (lane == 0) logits[n] = p;
        m = fmaxf(m, p);
    }

    float s = 0.f;
    for (int n = start + lane; n <= endi; n += 32)
        s += __expf(logits[n] - m);
    #pragma unroll
    for (int off = 16; off > 0; off >>= 1)
        s += __shfl_xor(s, off, 32);

    float inv = 1.0f / s;
    for (int n = start + lane; n <= endi; n += 32)
        out[n] = __expf(logits[n] - m) * inv;
}

// ---------------------------------------------------------------------------
extern "C" void kernel_launch(void* const* d_in, const int* in_sizes, int n_in,
                              void* d_out, int out_size, void* d_ws, size_t ws_size,
                              hipStream_t stream) {
    const float* X    = (const float*)d_in[0];   // [N, D]
    const int*   qend = (const int*)d_in[1];     // [N]
    const float* Wq   = (const float*)d_in[3];   // [A, D]
    const float* Wk   = (const float*)d_in[4];   // [A, D]
    float* out = (float*)d_out;

    // workspace layout (256B aligned slabs)
    char* ws = (char*)d_ws;
    size_t off = 0;
    auto alloc = [&](size_t bytes) { char* p = ws + off; off += (bytes + 255) & ~(size_t)255; return p; };
    int*    ends    = (int*)alloc(NSEG * sizeof(int));
    int*    bcnt    = (int*)alloc(1024 * sizeof(int));
    int*    boff    = (int*)alloc(1024 * sizeof(int));
    __bf16* Xe_bf   = (__bf16*)alloc((size_t)NSEG * DDIM * sizeof(__bf16));
    __bf16* Wq_bf   = (__bf16*)alloc((size_t)ADIM * DDIM * sizeof(__bf16));
    __bf16* WkT_bf  = (__bf16*)alloc((size_t)DDIM * ADIM * sizeof(__bf16));
    __bf16* Q_bf    = (__bf16*)alloc((size_t)NSEG * ADIM * sizeof(__bf16));
    float*  Vseg    = (float*)alloc((size_t)NSEG * DDIM * sizeof(float));
    float*  logits  = (float*)alloc((size_t)NTOK * sizeof(float));

    k_count <<<1024, 256, 0, stream>>>(qend, bcnt);
    k_scan  <<<1, 1, 0, stream>>>(bcnt, boff);
    k_ends  <<<1024, 256, 0, stream>>>(qend, boff, ends);
    k_prepw <<<ADIM, DDIM, 0, stream>>>(Wq, Wk, Wq_bf, WkT_bf);
    k_gather<<<NSEG, DDIM, 0, stream>>>(X, ends, Xe_bf);
    k_gemm_q<<<256, 256, 0, stream>>>(Xe_bf, Wq_bf, Q_bf);     // 2048 tiles
    k_gemm_v<<<512, 256, 0, stream>>>(Q_bf, WkT_bf, Vseg);     // 4096 tiles
    k_attend<<<512, 256, 0, stream>>>(X, ends, Vseg, logits, out);
}
